// ProbSparseAttention_23682449670169
// MI455X (gfx1250) — matrix operand order repaired and checked by reading, not compiled, not run on hardware
//
#include <hip/hip_runtime.h>
#include <hip/hip_bf16.h>
#include <math.h>

// ---------------------------------------------------------------------------
// ProbSparse attention on MI455X (gfx1250).
// B=8 S=4096 DM=1024 H=16 D=64, U=SAMPLE_K=45, SCALE=1/8.
// Heavy GEMMs in bf16 WMMA (16x16x32), fp32 accumulate.
// GEMM: 128x128 block tile double-buffered through LDS via
// GLOBAL_LOAD_ASYNC_TO_LDS_B128 (ASYNCcnt), 32x64 wave tile (8 WMMA / 6 frags).
// ---------------------------------------------------------------------------

#define B_   8
#define S_   4096
#define DM_  1024
#define H_   16
#define D_   64
#define U_   45
#define UPAD 48
#define SCALE_ 0.125f
#define PITCH 40   // LDS row pitch in bf16 (80B) -> staggers the 64 banks

typedef __bf16 bf16;
typedef __attribute__((ext_vector_type(16))) __bf16 v16bf;
typedef __attribute__((ext_vector_type(8)))  __bf16 v8bf;
typedef __attribute__((ext_vector_type(8)))  float  v8f;
typedef int v4i __attribute__((vector_size(16)));

#if defined(__HIP_DEVICE_COMPILE__) && \
    __has_builtin(__builtin_amdgcn_global_load_async_to_lds_b128) && \
    __has_builtin(__builtin_amdgcn_s_wait_asynccnt)
#define USE_ASYNC 1
typedef __attribute__((address_space(1))) v4i* gv4i_p;   // global int4*
typedef __attribute__((address_space(3))) v4i* lv4i_p;   // LDS    int4*
#else
#define USE_ASYNC 0
#endif

// --- WMMA helpers -----------------------------------------------------------

__device__ __forceinline__ v8f wmma_bf16(v16bf a, v16bf b, v8f c) {
  return __builtin_amdgcn_wmma_f32_16x16x32_bf16(false, a, false, b,
                                                 (short)0, c, false, false);
}

// A-matrix 16x32 fragment (row-major, leading dim ld, [0][0] at p0).
// lanes 0-15: row (lane&15), K = 0..7 / 16..23; lanes 16-31: K = 8..15 / 24..31.
__device__ __forceinline__ v16bf load_fragA(const bf16* __restrict__ p0, int ld, int lane) {
  const bf16* p = p0 + (size_t)(lane & 15) * ld + ((lane >> 4) << 3);
  v8bf lo = *(const v8bf*)p;
  v8bf hi = *(const v8bf*)(p + 16);
  return __builtin_shufflevector(lo, hi, 0,1,2,3,4,5,6,7,8,9,10,11,12,13,14,15);
}

__device__ __forceinline__ v16bf load_fragA_row(const bf16* __restrict__ rowp, int lane) {
  const bf16* p = rowp + ((lane >> 4) << 3);
  v8bf lo = *(const v8bf*)p;
  v8bf hi = *(const v8bf*)(p + 16);
  return __builtin_shufflevector(lo, hi, 0,1,2,3,4,5,6,7,8,9,10,11,12,13,14,15);
}

// B-matrix 32x16 fragment; column n of B = row n of a row-major [N][K] source
// (computes A @ W^T). lanes 0-15: col (lane&15), K = 0..7 / 16..23; lanes 16-31: K = 8..15 / 24..31.
__device__ __forceinline__ v16bf load_fragB(const bf16* __restrict__ p0, int ld, int lane) {
  const bf16* p = p0 + (size_t)(lane & 15) * ld + ((lane >> 4) << 3);
  v8bf lo = *(const v8bf*)p;
  v8bf hi = *(const v8bf*)(p + 16);
  return __builtin_shufflevector(lo, hi, 0,1,2,3,4,5,6,7,8,9,10,11,12,13,14,15);
}

__device__ __forceinline__ v16bf load_fragB_row(const bf16* __restrict__ rowp, int lane) {
  const bf16* p = rowp + ((lane >> 4) << 3);
  v8bf lo = *(const v8bf*)p;
  v8bf hi = *(const v8bf*)(p + 16);
  return __builtin_shufflevector(lo, hi, 0,1,2,3,4,5,6,7,8,9,10,11,12,13,14,15);
}

// --- async staging of a 128-row x 32-col bf16 tile into LDS ------------------
__device__ __forceinline__ void stage_tile(const bf16* __restrict__ gsrc, int ldg,
                                           bf16* lds, int k0, int tid) {
  // 128 rows x 64B, each thread moves 2 x 16B chunks (512 chunks total)
  for (int c = tid; c < 512; c += 256) {
    const int row = c >> 2, q = c & 3;
    const bf16* src = gsrc + (size_t)row * ldg + k0 + q * 8;
    bf16* dst = lds + row * PITCH + q * 8;
#if USE_ASYNC
    __builtin_amdgcn_global_load_async_to_lds_b128(
        (gv4i_p)(void*)src, (lv4i_p)(void*)dst, 0, 0);
#else
    *(v8bf*)dst = *(const v8bf*)src;
#endif
  }
}

// --- elementwise f32 -> bf16 ------------------------------------------------

__global__ void cvt_kernel(const float* __restrict__ src, bf16* __restrict__ dst, int n) {
  int i = blockIdx.x * 256 + threadIdx.x;
  if (i < n) dst[i] = (bf16)src[i];
}

// --- generic C = A(MxK) @ W(NxK)^T + bias, bf16 WMMA, store by MODE ----------
// MODE 0: bf16 out, (B,H,S,D) layout        (Q, K projections)
// MODE 1: bf16 out, (B,H,D,S) layout        (V projection, stored transposed)
// MODE 2: f32  out, row-major (M x N)       (final output projection)
// grid: (M/128, N/128); block 256 = 8 waves (4 M-groups x 2 N-groups),
// wave tile 32x64 (2x4 accum tiles); LDS double-buffered 32-deep K slices.
template <int MODE>
__global__ __launch_bounds__(256) void gemm_xWT(const bf16* __restrict__ A,
                                                const bf16* __restrict__ W,
                                                const float* __restrict__ bias,
                                                void* __restrict__ out,
                                                int K, int N) {
  __shared__ bf16 sA[2][128 * PITCH];
  __shared__ bf16 sW[2][128 * PITCH];

  const int tid  = threadIdx.x;
  const int lane = tid & 31;
  const int wid  = tid >> 5;
  const int wm   = wid >> 1;          // 0..3 -> 32-row group
  const int wn   = wid & 1;           // 0..1 -> 64-col group
  const int r0b  = blockIdx.x * 128;
  const int c0b  = blockIdx.y * 128;

  const bf16* Ab = A + (size_t)r0b * K;
  const bf16* Wb = W + (size_t)c0b * K;

  v8f acc[2][4] = {};
  const int NKS = K / 32;

  stage_tile(Ab, K, sA[0], 0, tid);
  stage_tile(Wb, K, sW[0], 0, tid);

  for (int ks = 0; ks < NKS; ++ks) {
    const int buf = ks & 1;
    if (ks + 1 < NKS) {
      stage_tile(Ab, K, sA[buf ^ 1], (ks + 1) * 32, tid);
      stage_tile(Wb, K, sW[buf ^ 1], (ks + 1) * 32, tid);
#if USE_ASYNC
      __builtin_amdgcn_s_wait_asynccnt(4);   // stage(ks) done, stage(ks+1) in flight
#endif
    } else {
#if USE_ASYNC
      __builtin_amdgcn_s_wait_asynccnt(0);
#endif
    }
    __syncthreads();

    const bf16* la = &sA[buf][wm * 32 * PITCH];
    const bf16* lw = &sW[buf][wn * 64 * PITCH];
    v16bf a0 = load_fragA(la, PITCH, lane);
    v16bf a1 = load_fragA(la + 16 * PITCH, PITCH, lane);
    v16bf b0 = load_fragB(lw, PITCH, lane);
    v16bf b1 = load_fragB(lw + 16 * PITCH, PITCH, lane);
    v16bf b2 = load_fragB(lw + 32 * PITCH, PITCH, lane);
    v16bf b3 = load_fragB(lw + 48 * PITCH, PITCH, lane);
    acc[0][0] = wmma_bf16(a0, b0, acc[0][0]);
    acc[0][1] = wmma_bf16(a0, b1, acc[0][1]);
    acc[0][2] = wmma_bf16(a0, b2, acc[0][2]);
    acc[0][3] = wmma_bf16(a0, b3, acc[0][3]);
    acc[1][0] = wmma_bf16(a1, b0, acc[1][0]);
    acc[1][1] = wmma_bf16(a1, b1, acc[1][1]);
    acc[1][2] = wmma_bf16(a1, b2, acc[1][2]);
    acc[1][3] = wmma_bf16(a1, b3, acc[1][3]);
    __syncthreads();   // protect buf from being restaged while still being read
  }

  // C/D layout: VGPR r, lanes 0-15 -> (M=r, N=lane); lanes 16-31 -> (M=r+8, N=lane-16)
  #pragma unroll
  for (int i = 0; i < 2; ++i) {
    #pragma unroll
    for (int j = 0; j < 4; ++j) {
      const int trow = r0b + wm * 32 + i * 16 + ((lane >> 4) << 3);
      const int ncol = c0b + wn * 64 + j * 16 + (lane & 15);
      #pragma unroll
      for (int r = 0; r < 8; ++r) {
        const int mrow = trow + r;
        float v = acc[i][j][r] + bias[ncol];
        if (MODE == 2) {
          ((float*)out)[(size_t)mrow * N + ncol] = v;
        } else {
          const int b = mrow >> 12, s = mrow & (S_ - 1);   // S = 4096
          const int h = ncol >> 6,  d = ncol & (D_ - 1);   // D = 64
          const size_t idx = (MODE == 0)
              ? ((((size_t)(b * H_ + h)) * S_ + s) * D_ + d)
              : ((((size_t)(b * H_ + h)) * D_ + d) * S_ + s);
          ((bf16*)out)[idx] = (bf16)v;
        }
      }
    }
  }
}

// --- sampled sparsity scores: M = max_k(QKs) - mean_k(QKs) -------------------
// grid: (S/128, B*H); block 256 = 8 waves; wave handles 16 queries x 48 cols.
__global__ void score_kernel(const bf16* __restrict__ Qb, const bf16* __restrict__ Kb,
                             const int* __restrict__ sidx, float* __restrict__ Msc) {
  const int lane = threadIdx.x & 31;
  const int wid  = threadIdx.x >> 5;
  const int bh   = blockIdx.y;
  const int s0   = blockIdx.x * 128 + wid * 16;
  const int myn  = lane & 15;

  const bf16* Qbase = Qb + (size_t)bh * S_ * D_;
  const bf16* Kbase = Kb + (size_t)bh * S_ * D_;

  int key[3];
  #pragma unroll
  for (int t = 0; t < 3; ++t) {
    int n = t * 16 + myn;
    key[t] = sidx[n < U_ ? n : (U_ - 1)];
  }

  v8f acc[3] = {};
  #pragma unroll
  for (int kk = 0; kk < 2; ++kk) {           // D = 64 -> 2 K-steps of 32
    v16bf a = load_fragA(Qbase + (size_t)s0 * D_ + kk * 32, D_, lane);
    #pragma unroll
    for (int t = 0; t < 3; ++t) {
      v16bf b = load_fragB_row(Kbase + (size_t)key[t] * D_ + kk * 32, lane);
      acc[t] = wmma_bf16(a, b, acc[t]);
    }
  }

  float rmax[8], rsum[8];
  #pragma unroll
  for (int r = 0; r < 8; ++r) {
    float m = -1e30f, s = 0.f;
    #pragma unroll
    for (int t = 0; t < 3; ++t) {
      int col = t * 16 + myn;
      float v = acc[t][r] * SCALE_;
      if (col < U_) { m = fmaxf(m, v); s += v; }
    }
    rmax[r] = m; rsum[r] = s;
  }
  #pragma unroll
  for (int off = 1; off < 16; off <<= 1) {
    #pragma unroll
    for (int r = 0; r < 8; ++r) {
      rmax[r] = fmaxf(rmax[r], __shfl_xor(rmax[r], off, 32));
      rsum[r] += __shfl_xor(rsum[r], off, 32);
    }
  }
  if (myn == 0) {   // lane 0 -> rows 0..7, lane 16 -> rows 8..15
    const int rowbase = s0 + ((lane >> 4) << 3);
    #pragma unroll
    for (int r = 0; r < 8; ++r)
      Msc[(size_t)bh * S_ + rowbase + r] = rmax[r] - rsum[r] * (1.0f / U_);
  }
}

// --- top-45 selection per (b,h): iterative block argmax over LDS -------------
__global__ void topk_kernel(const float* __restrict__ Msc, int* __restrict__ topIdx) {
  const int bh = blockIdx.x, tid = threadIdx.x;
  __shared__ float sm[S_];
  __shared__ float rv[256];
  __shared__ int   ri[256];
  const float* Mr = Msc + (size_t)bh * S_;
  for (int i = tid; i < S_; i += 256) sm[i] = Mr[i];
  __syncthreads();
  for (int u = 0; u < U_; ++u) {
    float best = -1e30f; int bi = 0;
    for (int i = tid; i < S_; i += 256) { float v = sm[i]; if (v > best) { best = v; bi = i; } }
    rv[tid] = best; ri[tid] = bi;
    __syncthreads();
    for (int off = 128; off; off >>= 1) {
      if (tid < off && rv[tid + off] > rv[tid]) { rv[tid] = rv[tid + off]; ri[tid] = ri[tid + off]; }
      __syncthreads();
    }
    if (tid == 0) { topIdx[bh * 64 + u] = ri[0]; sm[ri[0]] = -1e30f; }
    __syncthreads();
  }
}

// --- logits = Q_red @ K^T * SCALE -> (B,H,48,S) fp32 -------------------------
__global__ void qk_kernel(const bf16* __restrict__ Qb, const bf16* __restrict__ Kb,
                          const int* __restrict__ topIdx, float* __restrict__ Logits) {
  const int lane = threadIdx.x & 31;
  const int wid  = threadIdx.x >> 5;
  const int bh   = blockIdx.y;
  const int scol0 = blockIdx.x * 128 + wid * 16;
  const int myn  = lane & 15;

  const bf16* Qbase = Qb + (size_t)bh * S_ * D_;
  const bf16* Kbase = Kb + (size_t)bh * S_ * D_;
  const int*  ti    = topIdx + bh * 64;

  int qrow[3];
  #pragma unroll
  for (int mt = 0; mt < 3; ++mt) {
    int u = mt * 16 + myn;
    qrow[mt] = ti[u < U_ ? u : (U_ - 1)];
  }

  v8f acc[3] = {};
  #pragma unroll
  for (int kk = 0; kk < 2; ++kk) {
    v16bf b = load_fragB(Kbase + (size_t)scol0 * D_ + kk * 32, D_, lane);
    #pragma unroll
    for (int mt = 0; mt < 3; ++mt) {
      v16bf a = load_fragA_row(Qbase + (size_t)qrow[mt] * D_ + kk * 32, lane);
      acc[mt] = wmma_bf16(a, b, acc[mt]);
    }
  }

  float* L = Logits + (size_t)bh * UPAD * S_;
  #pragma unroll
  for (int mt = 0; mt < 3; ++mt) {
    const int u0 = mt * 16 + ((lane >> 4) << 3);
    #pragma unroll
    for (int r = 0; r < 8; ++r)
      L[(size_t)(u0 + r) * S_ + scol0 + myn] = acc[mt][r] * SCALE_;
  }
}

// --- row softmax over S=4096, write bf16 probabilities -----------------------
__global__ void softmax_kernel(const float* __restrict__ Logits, bf16* __restrict__ P) {
  const int row = blockIdx.x;            // bh*UPAD + u
  const int tid = threadIdx.x;
  const float* L = Logits + (size_t)row * S_;
  bf16* Pr = P + (size_t)row * S_;
  __shared__ float red[256];
  float m = -1e30f;
  for (int i = tid; i < S_; i += 256) m = fmaxf(m, L[i]);
  red[tid] = m; __syncthreads();
  for (int off = 128; off; off >>= 1) { if (tid < off) red[tid] = fmaxf(red[tid], red[tid + off]); __syncthreads(); }
  m = red[0]; __syncthreads();
  float s = 0.f;
  for (int i = tid; i < S_; i += 256) s += __expf(L[i] - m);
  red[tid] = s; __syncthreads();
  for (int off = 128; off; off >>= 1) { if (tid < off) red[tid] += red[tid + off]; __syncthreads(); }
  const float inv = 1.0f / red[0];
  for (int i = tid; i < S_; i += 256) Pr[i] = (bf16)(__expf(L[i] - m) * inv);
}

// --- out_red = P @ V  (V held transposed: rows of V^T are K-contiguous) ------
__global__ void pv_kernel(const bf16* __restrict__ P, const bf16* __restrict__ Vt,
                          float* __restrict__ OutRed) {
  const int lane = threadIdx.x & 31;
  const int wid  = threadIdx.x >> 5;
  const int bh   = blockIdx.x;
  const int mt = wid % 3, nt = wid / 3;

  const bf16* Pb = P  + (size_t)bh * UPAD * S_ + (size_t)(mt * 16) * S_;
  const bf16* Vb = Vt + (size_t)bh * D_   * S_ + (size_t)(nt * 16) * S_;

  v8f acc = {};
  for (int k0 = 0; k0 < S_; k0 += 32) {
    v16bf a = load_fragA(Pb + k0, S_, lane);
    v16bf b = load_fragB(Vb + k0, S_, lane);
    acc = wmma_bf16(a, b, acc);
  }
  float* O = OutRed + (size_t)bh * UPAD * D_;
  const int u0 = mt * 16 + ((lane >> 4) << 3);
  const int d  = nt * 16 + (lane & 15);
  #pragma unroll
  for (int r = 0; r < 8; ++r)
    O[(size_t)(u0 + r) * D_ + d] = acc[r];
}

// --- per-(b,h,d) mean over S of V (rows of V^T are contiguous) ---------------
__global__ void vmean_kernel(const bf16* __restrict__ Vt, float* __restrict__ Vmean) {
  const int row = blockIdx.x;            // bh*D + d
  const int tid = threadIdx.x;
  const bf16* Vr = Vt + (size_t)row * S_;
  __shared__ float red[256];
  float s = 0.f;
  for (int i = tid; i < S_; i += 256) s += (float)Vr[i];
  red[tid] = s; __syncthreads();
  for (int off = 128; off; off >>= 1) { if (tid < off) red[tid] += red[tid + off]; __syncthreads(); }
  if (tid == 0) Vmean[row] = red[0] * (1.0f / S_);
}

// --- fill ctx (B,S,DM) bf16 with broadcast V-mean ----------------------------
__global__ void fillctx_kernel(const float* __restrict__ Vmean, bf16* __restrict__ Ctx) {
  const size_t i = (size_t)blockIdx.x * 256 + threadIdx.x;   // < B*S*DM
  const int m = (int)(i & (DM_ - 1));
  const size_t bs = i >> 10;             // DM = 1024
  const int b = (int)(bs >> 12);         // S  = 4096
  const int h = m >> 6, d = m & 63;
  Ctx[i] = (bf16)Vmean[(b * H_ + h) * D_ + d];
}

// --- scatter out_red rows into ctx at top_idx positions ----------------------
__global__ void scatter_kernel(const float* __restrict__ OutRed,
                               const int* __restrict__ topIdx, bf16* __restrict__ Ctx) {
  const int bh = blockIdx.x;
  const int b = bh >> 4, h = bh & 15;
  const int* ti = topIdx + bh * 64;
  const float* O = OutRed + (size_t)bh * UPAD * D_;
  for (int t = threadIdx.x; t < U_ * D_; t += blockDim.x) {
    const int u = t >> 6, d = t & 63;
    const int s = ti[u];
    Ctx[((size_t)b * S_ + s) * DM_ + (h << 6) + d] = (bf16)O[u * D_ + d];
  }
}

// ---------------------------------------------------------------------------

extern "C" void kernel_launch(void* const* d_in, const int* in_sizes, int n_in,
                              void* d_out, int out_size, void* d_ws, size_t ws_size,
                              hipStream_t stream) {
  (void)in_sizes; (void)n_in; (void)out_size; (void)ws_size;
  const float* x   = (const float*)d_in[0];
  const int*  sidx = (const int*)  d_in[1];
  const float* Wq  = (const float*)d_in[2];
  const float* bq  = (const float*)d_in[3];
  const float* Wk  = (const float*)d_in[4];
  const float* bk  = (const float*)d_in[5];
  const float* Wv  = (const float*)d_in[6];
  const float* bv  = (const float*)d_in[7];
  const float* Wo  = (const float*)d_in[8];
  const float* bo  = (const float*)d_in[9];

  char* ws = (char*)d_ws;
  size_t off = 0;
  auto alloc = [&](size_t bytes) { char* p = ws + off; off += (bytes + 255) & ~(size_t)255; return p; };

  const size_t NX = (size_t)B_ * S_ * DM_;        // 33,554,432
  const size_t NW = (size_t)DM_ * DM_;            // 1,048,576
  const size_t NQ = (size_t)B_ * H_ * S_ * D_;    // == NX

  bf16*  XB   = (bf16*) alloc(NX * 2);            // bf16 x; later reused as Ctx
  bf16*  WQB  = (bf16*) alloc(NW * 2);
  bf16*  WKB  = (bf16*) alloc(NW * 2);
  bf16*  WVB  = (bf16*) alloc(NW * 2);
  bf16*  WOB  = (bf16*) alloc(NW * 2);
  bf16*  Qb   = (bf16*) alloc(NQ * 2);            // (B,H,S,D)
  bf16*  Kb   = (bf16*) alloc(NQ * 2);            // (B,H,S,D)
  bf16*  Vtb  = (bf16*) alloc(NQ * 2);            // (B,H,D,S)
  float* Msc  = (float*)alloc((size_t)B_ * H_ * S_ * 4);
  int*   TIdx = (int*)  alloc((size_t)B_ * H_ * 64 * 4);
  float* Lg   = (float*)alloc((size_t)B_ * H_ * UPAD * S_ * 4);
  bf16*  Pm   = (bf16*) alloc((size_t)B_ * H_ * UPAD * S_ * 2);
  float* ORed = (float*)alloc((size_t)B_ * H_ * UPAD * D_ * 4);
  float* Vmn  = (float*)alloc((size_t)B_ * H_ * D_ * 4);
  bf16*  Ctx  = XB;                               // x no longer needed after projections

  // 1) fp32 -> bf16 conversions
  cvt_kernel<<<(unsigned)((NX + 255) / 256), 256, 0, stream>>>(x,  XB,  (int)NX);
  cvt_kernel<<<(unsigned)((NW + 255) / 256), 256, 0, stream>>>(Wq, WQB, (int)NW);
  cvt_kernel<<<(unsigned)((NW + 255) / 256), 256, 0, stream>>>(Wk, WKB, (int)NW);
  cvt_kernel<<<(unsigned)((NW + 255) / 256), 256, 0, stream>>>(Wv, WVB, (int)NW);
  cvt_kernel<<<(unsigned)((NW + 255) / 256), 256, 0, stream>>>(Wo, WOB, (int)NW);

  // 2) Q/K/V projections (M=32768, N=1024, K=1024)
  dim3 ggrid(B_ * S_ / 128, DM_ / 128);
  gemm_xWT<0><<<ggrid, 256, 0, stream>>>(XB, WQB, bq, Qb,  DM_, DM_);
  gemm_xWT<0><<<ggrid, 256, 0, stream>>>(XB, WKB, bk, Kb,  DM_, DM_);
  gemm_xWT<1><<<ggrid, 256, 0, stream>>>(XB, WVB, bv, Vtb, DM_, DM_);

  // 3) sampled sparsity scores + top-45 queries
  score_kernel<<<dim3(S_ / 128, B_ * H_), 256, 0, stream>>>(Qb, Kb, sidx, Msc);
  topk_kernel<<<B_ * H_, 256, 0, stream>>>(Msc, TIdx);

  // 4) reduced attention: logits, softmax, P @ V
  qk_kernel<<<dim3(S_ / 128, B_ * H_), 256, 0, stream>>>(Qb, Kb, TIdx, Lg);
  softmax_kernel<<<B_ * H_ * UPAD, 256, 0, stream>>>(Lg, Pm);
  pv_kernel<<<B_ * H_, 384, 0, stream>>>(Pm, Vtb, ORed);

  // 5) compose context: V-mean base + scattered reduced outputs
  vmean_kernel<<<B_ * H_ * D_, 256, 0, stream>>>(Vtb, Vmn);
  fillctx_kernel<<<(unsigned)(NX / 256), 256, 0, stream>>>(Vmn, Ctx);
  scatter_kernel<<<B_ * H_, 256, 0, stream>>>(ORed, TIdx, Ctx);

  // 6) output projection -> d_out fp32 (B,S,DM)
  gemm_xWT<2><<<ggrid, 256, 0, stream>>>(Ctx, WOB, bo, (float*)d_out, DM_, DM_);
}
